// Encoder_41386304864522
// MI455X (gfx1250) — compile-verified, matrix operand order checked by reading, not run on hardware
//
#include <hip/hip_runtime.h>
#include <hip/hip_bf16.h>
#include <hip/hip_fp16.h>

// Problem dims (fixed by reference)
#define NB     4
#define LL     1024
#define HH     1024
#define NHEADS 16
#define HS     64
#define MTOT   (NB * LL)
#define CHUNK  256   // kv tokens staged in LDS per iteration

typedef __attribute__((ext_vector_type(16))) _Float16 v16h;
typedef __attribute__((ext_vector_type(8)))  _Float16 v8h;
typedef __attribute__((ext_vector_type(8)))  float    v8f;
typedef __attribute__((ext_vector_type(4)))  int      v4i;

#define AS1 __attribute__((address_space(1)))
#define AS3 __attribute__((address_space(3)))

#if __has_builtin(__builtin_amdgcn_global_load_async_to_lds_b128)
#define HAVE_ASYNC_LDS 1
#endif

union AFrag { v16h v; v8h h[2]; };

static __device__ inline v8f zero8() {
    v8f z;
#pragma unroll
    for (int i = 0; i < 8; ++i) z[i] = 0.0f;
    return z;
}

// Load a 16x32 (A) or 32x16-as-Bt (B) f16 WMMA fragment from a row-major
// matrix. ISA layout: lane m = lane&15 holds row/col m; kbase = 8*(lane>>4);
// vector elems 0..7 = K kb..kb+7, elems 8..15 = K kb+16..kb+23.
static __device__ inline v16h load_frag16(const _Float16* base, int ld, int lane) {
    int m  = lane & 15;
    int kb = (lane >> 4) << 3;
    const _Float16* p = base + (size_t)m * ld + kb;
    AFrag f;
    f.h[0] = *(const v8h*)(p);
    f.h[1] = *(const v8h*)(p + 16);
    return f.v;
}

static __device__ inline v8f wmma16(v16h a, v16h b, v8f c) {
    return __builtin_amdgcn_wmma_f32_16x16x32_f16(
        /*neg_a=*/false, a, /*neg_b=*/false, b,
        /*c_mod=*/(short)0, c, /*reuse_a=*/false, /*reuse_b=*/false);
}

// 16-byte global -> LDS copy, async on CDNA5 when the builtin is available.
static __device__ inline void stage16(const _Float16* g, _Float16* l) {
#ifdef HAVE_ASYNC_LDS
    __builtin_amdgcn_global_load_async_to_lds_b128(
        (AS1 v4i*)(const_cast<_Float16*>(g)), (AS3 v4i*)l, 0, 0);
#else
    *(v8h*)l = *(const v8h*)g;
#endif
}

static __device__ inline void wait_stage() {
#ifdef HAVE_ASYNC_LDS
#if __has_builtin(__builtin_amdgcn_s_wait_asynccnt)
    __builtin_amdgcn_s_wait_asynccnt(0);
#else
    asm volatile("s_wait_asynccnt 0" ::: "memory");
#endif
#endif
}

// ---------------------------------------------------------------------------
// f32 -> f16 conversion
// ---------------------------------------------------------------------------
__global__ __launch_bounds__(256) void f32_to_f16(const float* __restrict__ s,
                                                  _Float16* __restrict__ d, int n) {
    int i = blockIdx.x * 256 + threadIdx.x;
    if (i < n) d[i] = (_Float16)s[i];
}

// ---------------------------------------------------------------------------
// y = A (M x 1024, f16) @ W^T (+bias). W given row-major (n,k) f16, which is
// exactly the B-fragment source layout. One wave computes a 64x64 tile.
// mode 0: store f16, (row, col) row-major (Q, K projections)
// mode 1: store f16 transposed per head: vt[((n*NHEADS+h)*HS+s)*LL + l]
// mode 2: store f32 row-major (final output)
// ---------------------------------------------------------------------------
__global__ __launch_bounds__(32) void gemm_qkvo(const _Float16* __restrict__ A,
                                                const _Float16* __restrict__ W,
                                                const float* __restrict__ bias,
                                                void* __restrict__ out, int mode) {
    int lane  = threadIdx.x;
    int nbase = blockIdx.x * 64;
    int mbase = blockIdx.y * 64;

    v8f acc[4][4];
#pragma unroll
    for (int mt = 0; mt < 4; ++mt)
#pragma unroll
        for (int nt = 0; nt < 4; ++nt) acc[mt][nt] = zero8();

    int m0 = lane & 15;
    int kb = (lane >> 4) << 3;

#pragma unroll 1
    for (int kk = 0; kk < HH; kk += 32) {
        // Pull next K-step's lines toward the WGP while this step computes.
        __builtin_prefetch(A + (size_t)(mbase + m0) * HH + kk + 64 + kb, 0, 3);
        __builtin_prefetch(W + (size_t)(nbase + m0) * HH + kk + 64 + kb, 0, 3);

        v16h a[4], b[4];
#pragma unroll
        for (int mt = 0; mt < 4; ++mt)
            a[mt] = load_frag16(A + (size_t)(mbase + 16 * mt) * HH + kk, HH, lane);
#pragma unroll
        for (int nt = 0; nt < 4; ++nt)
            b[nt] = load_frag16(W + (size_t)(nbase + 16 * nt) * HH + kk, HH, lane);
#pragma unroll
        for (int mt = 0; mt < 4; ++mt)
#pragma unroll
            for (int nt = 0; nt < 4; ++nt)
                acc[mt][nt] = wmma16(a[mt], b[nt], acc[mt][nt]);
    }

    int r0 = (lane >> 4) << 3;   // 0 or 8
    int cc = lane & 15;
#pragma unroll
    for (int mt = 0; mt < 4; ++mt) {
#pragma unroll
        for (int nt = 0; nt < 4; ++nt) {
            int col  = nbase + nt * 16 + cc;
            float bv = bias[col];
#pragma unroll
            for (int g = 0; g < 8; ++g) {
                float val = acc[mt][nt][g] + bv;
                int row   = mbase + mt * 16 + r0 + g;
                if (mode == 0) {
                    ((_Float16*)out)[(size_t)row * HH + col] = (_Float16)val;
                } else if (mode == 1) {
                    int n = row >> 10, l = row & 1023;
                    int h = col >> 6, s = col & 63;
                    ((_Float16*)out)[(((size_t)((n * NHEADS + h) * HS + s)) << 10) + l] =
                        (_Float16)val;
                } else {
                    ((float*)out)[(size_t)row * HH + col] = val;
                }
            }
        }
    }
}

// ---------------------------------------------------------------------------
// Fused flash attention for one (n, h, q-group of 128 rows).
// 8 waves/block; each wave owns a 16-row query block. K chunk and V^T chunk
// staged in LDS (async global->LDS when available) and shared by all waves;
// online softmax; P goes through a per-wave LDS staging tile to re-layout
// C-tile -> A-fragment.
// ---------------------------------------------------------------------------
__global__ __launch_bounds__(256) void attn_fused(const _Float16* __restrict__ qh,
                                                  const _Float16* __restrict__ kh,
                                                  const _Float16* __restrict__ vth,
                                                  _Float16* __restrict__ ctx) {
    __shared__ __align__(16) _Float16 Ksh[CHUNK][HS];     // 32 KB
    __shared__ __align__(16) _Float16 VshT[HS][CHUNK];    // 32 KB
    __shared__ __align__(16) _Float16 Pst[8][16][32];     // 8 KB

    int tid  = threadIdx.x;
    int lane = tid & 31;
    int wv   = tid >> 5;
    int hIdx = blockIdx.y;
    int nIdx = blockIdx.z;
    int q0   = blockIdx.x * 128 + wv * 16;

    // Q fragments for this wave's 16 query rows (constant across kv loop)
    const _Float16* qp = qh + ((size_t)(nIdx * LL + q0)) * HH + hIdx * HS;
    v16h qa0 = load_frag16(qp, HH, lane);        // s = 0..31
    v16h qa1 = load_frag16(qp + 32, HH, lane);   // s = 32..63

    float mrow[8], lrow[8];
    v8f o[4];
#pragma unroll
    for (int g = 0; g < 8; ++g) { mrow[g] = -3.0e38f; lrow[g] = 0.0f; }
#pragma unroll
    for (int st = 0; st < 4; ++st) o[st] = zero8();

    int r0 = (lane >> 4) << 3;
    int cc = lane & 15;

#pragma unroll 1
    for (int c0 = 0; c0 < LL; c0 += CHUNK) {
        // Cooperative stage: K chunk (row-major, 64-wide rows)
        for (int idx = tid; idx < CHUNK * (HS / 8); idx += 256) {
            int t = idx >> 3, part = idx & 7;
            stage16(kh + ((size_t)(nIdx * LL + c0 + t)) * HH + hIdx * HS + part * 8,
                    &Ksh[t][part * 8]);
        }
        // Cooperative stage: V^T chunk (vth is already per-head transposed)
        for (int idx = tid; idx < HS * (CHUNK / 8); idx += 256) {
            int s = idx >> 5, part = idx & 31;   // CHUNK/8 == 32
            stage16(vth + (((size_t)((nIdx * NHEADS + hIdx) * HS + s)) << 10) + c0 + part * 8,
                    &VshT[s][part * 8]);
        }
        wait_stage();
        __syncthreads();

#pragma unroll 2
        for (int j = 0; j < CHUNK; j += 32) {
            // S = Q @ K^T for 32 keys (two 16-key tiles), K-dim = 64 (2 steps)
            v16h kb0l = load_frag16(&Ksh[j][0], HS, lane);
            v16h kb0h = load_frag16(&Ksh[j][32], HS, lane);
            v16h kb1l = load_frag16(&Ksh[j + 16][0], HS, lane);
            v16h kb1h = load_frag16(&Ksh[j + 16][32], HS, lane);
            v8f s0 = zero8(), s1 = zero8();
            s0 = wmma16(qa0, kb0l, s0);
            s0 = wmma16(qa1, kb0h, s0);
            s1 = wmma16(qa0, kb1l, s1);
            s1 = wmma16(qa1, kb1h, s1);

            // Online softmax: row max over 32 keys (16 lanes x 2 tiles)
            float mx[8];
#pragma unroll
            for (int g = 0; g < 8; ++g) mx[g] = fmaxf(s0[g], s1[g]);
#pragma unroll
            for (int off = 1; off < 16; off <<= 1)
#pragma unroll
                for (int g = 0; g < 8; ++g)
                    mx[g] = fmaxf(mx[g], __shfl_xor(mx[g], off, 32));

            float p0[8], p1[8], sc[8], rs[8];
#pragma unroll
            for (int g = 0; g < 8; ++g) {
                float mn = fmaxf(mrow[g], mx[g]);
                sc[g]    = __expf(mrow[g] - mn);
                mrow[g]  = mn;
                p0[g]    = __expf(s0[g] - mn);
                p1[g]    = __expf(s1[g] - mn);
                rs[g]    = p0[g] + p1[g];
            }
#pragma unroll
            for (int off = 1; off < 16; off <<= 1)
#pragma unroll
                for (int g = 0; g < 8; ++g) rs[g] += __shfl_xor(rs[g], off, 32);
#pragma unroll
            for (int g = 0; g < 8; ++g) lrow[g] = lrow[g] * sc[g] + rs[g];
#pragma unroll
            for (int st = 0; st < 4; ++st)
#pragma unroll
                for (int g = 0; g < 8; ++g) o[st][g] *= sc[g];

            // C-layout -> A-fragment relayout of P through per-wave LDS tile
#pragma unroll
            for (int g = 0; g < 8; ++g) {
                Pst[wv][r0 + g][cc]      = (_Float16)p0[g];
                Pst[wv][r0 + g][cc + 16] = (_Float16)p1[g];
            }
            asm volatile("s_wait_dscnt 0" ::: "memory");
            v16h pa = load_frag16(&Pst[wv][0][0], 32, lane);

            // o += P (16x32) @ V (32x64): 4 N-tiles of 16 along head dim
#pragma unroll
            for (int st = 0; st < 4; ++st) {
                v16h vb = load_frag16(&VshT[st * 16][j], CHUNK, lane);
                o[st]   = wmma16(pa, vb, o[st]);
            }
        }
        __syncthreads();
    }

    // Normalize and store ctx (f16, standard (n,l,h*64+s) layout)
#pragma unroll
    for (int st = 0; st < 4; ++st) {
#pragma unroll
        for (int g = 0; g < 8; ++g) {
            float val = o[st][g] / lrow[g];
            int q     = q0 + r0 + g;
            ctx[((size_t)(nIdx * LL + q)) * HH + hIdx * HS + st * 16 + cc] = (_Float16)val;
        }
    }
}

// ---------------------------------------------------------------------------
// Host-side orchestration
// ---------------------------------------------------------------------------
extern "C" void kernel_launch(void* const* d_in, const int* in_sizes, int n_in,
                              void* d_out, int out_size, void* d_ws, size_t ws_size,
                              hipStream_t stream) {
    (void)in_sizes; (void)n_in; (void)out_size; (void)ws_size;

    const float* x  = (const float*)d_in[0];
    const float* Wq = (const float*)d_in[1];
    const float* bq = (const float*)d_in[2];
    const float* Wk = (const float*)d_in[3];
    const float* bk = (const float*)d_in[4];
    const float* Wv = (const float*)d_in[5];
    const float* bv = (const float*)d_in[6];
    const float* Wo = (const float*)d_in[7];
    const float* bo = (const float*)d_in[8];

    char* ws = (char*)d_ws;
    _Float16* xh   = (_Float16*)(ws);                          // 8 MB
    _Float16* wqh  = (_Float16*)(ws + ((size_t)8  << 20));     // 2 MB
    _Float16* wkh  = (_Float16*)(ws + ((size_t)10 << 20));     // 2 MB
    _Float16* wvh  = (_Float16*)(ws + ((size_t)12 << 20));     // 2 MB
    _Float16* woh  = (_Float16*)(ws + ((size_t)14 << 20));     // 2 MB
    _Float16* qhp  = (_Float16*)(ws + ((size_t)16 << 20));     // 8 MB
    _Float16* khp  = (_Float16*)(ws + ((size_t)24 << 20));     // 8 MB
    _Float16* vtp  = (_Float16*)(ws + ((size_t)32 << 20));     // 8 MB
    _Float16* ctxh = (_Float16*)(ws + ((size_t)40 << 20));     // 8 MB

    const int NX = MTOT * HH;      // 4M elements
    const int NW = HH * HH;        // 1M elements

    f32_to_f16<<<(NX + 255) / 256, 256, 0, stream>>>(x, xh, NX);
    f32_to_f16<<<(NW + 255) / 256, 256, 0, stream>>>(Wq, wqh, NW);
    f32_to_f16<<<(NW + 255) / 256, 256, 0, stream>>>(Wk, wkh, NW);
    f32_to_f16<<<(NW + 255) / 256, 256, 0, stream>>>(Wv, wvh, NW);
    f32_to_f16<<<(NW + 255) / 256, 256, 0, stream>>>(Wo, woh, NW);

    dim3 gg(HH / 64, MTOT / 64);   // (16, 64)
    gemm_qkvo<<<gg, 32, 0, stream>>>(xh, wqh, bq, qhp, 0);
    gemm_qkvo<<<gg, 32, 0, stream>>>(xh, wkh, bk, khp, 0);
    gemm_qkvo<<<gg, 32, 0, stream>>>(xh, wvh, bv, vtp, 1);

    attn_fused<<<dim3(LL / 128, NHEADS, NB), 256, 0, stream>>>(qhp, khp, vtp, ctxh);

    gemm_qkvo<<<gg, 32, 0, stream>>>(ctxh, woh, bo, d_out, 2);
}
